// DistSAGE_70471823393514
// MI455X (gfx1250) — matrix operand order repaired and checked by reading, not compiled, not run on hardware
//
#include <hip/hip_runtime.h>

typedef __attribute__((ext_vector_type(16))) _Float16 v16h;
typedef __attribute__((ext_vector_type(8)))  _Float16 v8h;
typedef __attribute__((ext_vector_type(4)))  _Float16 h4;
typedef __attribute__((ext_vector_type(8)))  float    v8f;
typedef __attribute__((ext_vector_type(4)))  float    f4v;

// vector-of-4-int pointee types matching the async-to-LDS builtin signature
typedef int v4i __attribute__((vector_size(16)));
typedef __attribute__((address_space(1))) v4i gv4i;
typedef __attribute__((address_space(3))) v4i lv4i;

#define LDS_STRIDE 40   // 32 halves payload + 8 halves pad -> 80B rows, 16B aligned

#if defined(__gfx1250__) &&                                             \
    __has_builtin(__builtin_amdgcn_global_load_async_to_lds_b128) &&    \
    __has_builtin(__builtin_amdgcn_s_wait_asynccnt)
#define USE_ASYNC_LDS 1
#else
#define USE_ASYNC_LDS 0
#endif

// ---------------------------------------------------------------------------
// Weight pack: Wt[n][k] (row-major, f16), k < din -> Wself[k][n], else Wneigh.
// Rows n >= dout are zero-padded (used to pad D_OUT=19 -> 64).
// ---------------------------------------------------------------------------
__global__ void pack_wt_kernel(const float* __restrict__ Wself,
                               const float* __restrict__ Wneigh,
                               _Float16* __restrict__ Wt,
                               int din, int dout, int npad) {
    int idx = blockIdx.x * blockDim.x + threadIdx.x;
    int K2 = 2 * din;
    if (idx >= npad * K2) return;
    int n = idx / K2;
    int k = idx - n * K2;
    float v = 0.f;
    if (n < dout)
        v = (k < din) ? Wself[(size_t)k * dout + n]
                      : Wneigh[(size_t)(k - din) * dout + n];
    Wt[idx] = (_Float16)v;
}

__global__ void pad_bias_kernel(const float* __restrict__ b,
                                float* __restrict__ bpad, int n, int npad) {
    int i = threadIdx.x;
    if (i < npad) bpad[i] = (i < n) ? b[i] : 0.f;
}

// ---------------------------------------------------------------------------
// Layer 0 activation build: A0[row] = [ f16(x[row]) | f16(mean_j x[nbr[row][j]]) ]
// One block per dst row, 256 threads * float4 = 1024 cols. Gather loads are
// non-temporal: x footprint (~1 GB) >> L2 (192 MB), expected reuse ~1.
// ---------------------------------------------------------------------------
__global__ void build_a0_kernel(const float* __restrict__ x,
                                const int* __restrict__ nbr,
                                _Float16* __restrict__ A) {
    const int row = blockIdx.x;
    const int t   = threadIdx.x;              // 0..255, 4 f32 cols each
    _Float16* arow = A + (size_t)row * 2048;

    // self half
    const f4v s = ((const f4v*)(x + (size_t)row * 1024))[t];
    h4 hs; hs[0] = (_Float16)s[0]; hs[1] = (_Float16)s[1];
           hs[2] = (_Float16)s[2]; hs[3] = (_Float16)s[3];
    ((h4*)arow)[t] = hs;

    // neighbor mean half (streaming loads)
    float a0 = 0.f, a1 = 0.f, a2 = 0.f, a3 = 0.f;
    const int* nb = nbr + row * 10;
    #pragma unroll
    for (int j = 0; j < 10; ++j) {
        const f4v v = __builtin_nontemporal_load(
            ((const f4v*)(x + (size_t)nb[j] * 1024)) + t);
        a0 += v[0]; a1 += v[1]; a2 += v[2]; a3 += v[3];
    }
    const float inv = 0.1f;
    h4 hn; hn[0] = (_Float16)(a0 * inv); hn[1] = (_Float16)(a1 * inv);
           hn[2] = (_Float16)(a2 * inv); hn[3] = (_Float16)(a3 * inv);
    ((h4*)(arow + 1024))[t] = hn;
}

// ---------------------------------------------------------------------------
// Generic f16 activation build for layers 1/2 (d = 256, blockDim = d/4 = 64).
// ---------------------------------------------------------------------------
__global__ void build_a_f16_kernel(const _Float16* __restrict__ h,
                                   const int* __restrict__ nbr,
                                   _Float16* __restrict__ A,
                                   int d, int fanout, float invf) {
    const int row = blockIdx.x;
    const int t   = threadIdx.x;              // d/4 threads
    _Float16* arow = A + (size_t)row * (2 * d);

    const h4 s = ((const h4*)(h + (size_t)row * d))[t];
    ((h4*)arow)[t] = s;

    float a0 = 0.f, a1 = 0.f, a2 = 0.f, a3 = 0.f;
    const int* nb = nbr + row * fanout;
    for (int j = 0; j < fanout; ++j) {
        const h4 v = ((const h4*)(h + (size_t)nb[j] * d))[t];
        a0 += (float)v[0]; a1 += (float)v[1];
        a2 += (float)v[2]; a3 += (float)v[3];
    }
    h4 hn; hn[0] = (_Float16)(a0 * invf); hn[1] = (_Float16)(a1 * invf);
           hn[2] = (_Float16)(a2 * invf); hn[3] = (_Float16)(a3 * invf);
    ((h4*)(arow + d))[t] = hn;
}

// ---------------------------------------------------------------------------
// WMMA GEMM: C[M][ldc] = act(A[M][K] @ Bt[Npad][K]^T + bias)
//   block tile 128(M) x 64(N), 8 waves (4x2), each wave 32x32 via 2x2
//   v_wmma_f32_16x16x32_f16 accumulating f32.
//   Double-buffered LDS; tiles staged with GLOBAL_LOAD_ASYNC_TO_LDS_B128
//   (ASYNCcnt) when available, bypassing VGPRs and overlapping with WMMA.
// ---------------------------------------------------------------------------
template<bool RELU, bool OUT_HALF>
__global__ __launch_bounds__(256)
void gemm_wmma_kernel(const _Float16* __restrict__ A,
                      const _Float16* __restrict__ Bt,
                      const float* __restrict__ bias,
                      void* __restrict__ Cv,
                      int K, int ldc, int ncols) {
    __shared__ _Float16 As[2][128 * LDS_STRIDE];
    __shared__ _Float16 Bs[2][64 * LDS_STRIDE];

    const int tid  = threadIdx.x;
    const int lane = tid & 31;
    const int w    = tid >> 5;
    const int wm   = w >> 1;        // 0..3 -> 32-row slab
    const int wn   = w & 1;         // 0..1 -> 32-col slab
    const int m16  = lane & 15;
    const int hi   = lane >> 4;     // lane half
    const int row0 = blockIdx.x * 128;
    const int col0 = blockIdx.y * 64;

    // Per-thread staging coordinates (16B chunks)
    const int ra = tid >> 2;                 // A rows for chunk 0 / +64 for chunk 1
    const int oa = (tid & 3) * 8;            // half offset within 32-half row
    const int rb = tid >> 2;                 // B row
    const int ob = (tid & 3) * 8;

    auto stage = [&](int buf, int k0) {
        // A tile: 128 rows x 32 halves -> 512 x 16B chunks, 2 per thread
        #pragma unroll
        for (int c = 0; c < 2; ++c) {
            const int r = ra + c * 64;
            const _Float16* g = A + (size_t)(row0 + r) * K + k0 + oa;
            _Float16* l = &As[buf][r * LDS_STRIDE + oa];
#if USE_ASYNC_LDS
            __builtin_amdgcn_global_load_async_to_lds_b128(
                (gv4i*)g, (lv4i*)l, 0, 0);
#else
            *(uint4*)l = *(const uint4*)g;
#endif
        }
        // B tile: 64 rows x 32 halves -> 256 x 16B chunks, 1 per thread
        {
            const _Float16* g = Bt + (size_t)(col0 + rb) * K + k0 + ob;
            _Float16* l = &Bs[buf][rb * LDS_STRIDE + ob];
#if USE_ASYNC_LDS
            __builtin_amdgcn_global_load_async_to_lds_b128(
                (gv4i*)g, (lv4i*)l, 0, 0);
#else
            *(uint4*)l = *(const uint4*)g;
#endif
        }
    };

    v8f acc[2][2];
    #pragma unroll
    for (int i = 0; i < 2; ++i)
        #pragma unroll
        for (int j = 0; j < 2; ++j)
            #pragma unroll
            for (int e = 0; e < 8; ++e) acc[i][j][e] = 0.f;

    stage(0, 0);

    int it = 0;
    for (int k0 = 0; k0 < K; k0 += 32, ++it) {
        const int buf = it & 1;
#if USE_ASYNC_LDS
        __builtin_amdgcn_s_wait_asynccnt(0);
#endif
        __syncthreads();   // staged tile visible; prev iteration's reads done
        if (k0 + 32 < K) stage(buf ^ 1, k0 + 32);

        v16h af[2], bf[2];
        // A fragment: lane half 'hi' -> k in {hi*8..+7} and {16+hi*8..+7}
        #pragma unroll
        for (int mi = 0; mi < 2; ++mi) {
            const _Float16* p = &As[buf][(wm * 32 + mi * 16 + m16) * LDS_STRIDE];
            v8h lo  = *(const v8h*)(p + hi * 8);
            v8h hi8 = *(const v8h*)(p + 16 + hi * 8);
            #pragma unroll
            for (int e = 0; e < 8; ++e) { af[mi][e] = lo[e]; af[mi][8 + e] = hi8[e]; }
        }
        // B fragment: lane half 'hi' -> k in {hi*8..+7} and {16+hi*8..+7}
        #pragma unroll
        for (int ni = 0; ni < 2; ++ni) {
            const _Float16* p = &Bs[buf][(wn * 32 + ni * 16 + m16) * LDS_STRIDE];
            v8h lo  = *(const v8h*)(p + hi * 8);
            v8h hi8 = *(const v8h*)(p + 16 + hi * 8);
            #pragma unroll
            for (int e = 0; e < 8; ++e) { bf[ni][e] = lo[e]; bf[ni][8 + e] = hi8[e]; }
        }

        #pragma unroll
        for (int mi = 0; mi < 2; ++mi)
            #pragma unroll
            for (int ni = 0; ni < 2; ++ni)
                acc[mi][ni] = __builtin_amdgcn_wmma_f32_16x16x32_f16(
                    false, af[mi], false, bf[ni],
                    (short)0, acc[mi][ni], false, false);
    }

    // Epilogue: D layout -> lane (n = m16), vgpr e -> row = 8*hi + e
    #pragma unroll
    for (int mi = 0; mi < 2; ++mi) {
        #pragma unroll
        for (int ni = 0; ni < 2; ++ni) {
            const int col  = col0 + wn * 32 + ni * 16 + m16;
            const float bc = bias[col];
            #pragma unroll
            for (int e = 0; e < 8; ++e) {
                const int row = row0 + wm * 32 + mi * 16 + hi * 8 + e;
                float v = acc[mi][ni][e] + bc;
                if (RELU) v = fmaxf(v, 0.f);
                if (OUT_HALF) {
                    ((_Float16*)Cv)[(size_t)row * ldc + col] = (_Float16)v;
                } else {
                    if (col < ncols)
                        ((float*)Cv)[(size_t)row * ldc + col] = v;
                }
            }
        }
    }
}

// ---------------------------------------------------------------------------
extern "C" void kernel_launch(void* const* d_in, const int* in_sizes, int n_in,
                              void* d_out, int out_size, void* d_ws, size_t ws_size,
                              hipStream_t stream) {
    (void)in_sizes; (void)n_in; (void)out_size; (void)ws_size;

    const float* x       = (const float*)d_in[0];
    const int*   nbr0    = (const int*)d_in[1];
    const int*   nbr1    = (const int*)d_in[2];
    const int*   nbr2    = (const int*)d_in[3];
    const float* Wself0  = (const float*)d_in[4];
    const float* Wneigh0 = (const float*)d_in[5];
    const float* b0      = (const float*)d_in[6];
    const float* Wself1  = (const float*)d_in[7];
    const float* Wneigh1 = (const float*)d_in[8];
    const float* b1      = (const float*)d_in[9];
    const float* Wself2  = (const float*)d_in[10];
    const float* Wneigh2 = (const float*)d_in[11];
    const float* b2      = (const float*)d_in[12];

    char* ws = (char*)d_ws;
    size_t off = 0;
    auto alloc = [&](size_t bytes) -> char* {
        char* p = ws + off;
        off += (bytes + 255) & ~(size_t)255;
        return p;
    };
    _Float16* W0t = (_Float16*)alloc(256ull * 2048 * 2);
    _Float16* W1t = (_Float16*)alloc(256ull * 512 * 2);
    _Float16* W2t = (_Float16*)alloc(64ull * 512 * 2);
    float*    b2p = (float*)   alloc(64 * sizeof(float));
    _Float16* A0  = (_Float16*)alloc(25600ull * 2048 * 2);
    _Float16* h1  = (_Float16*)alloc(25600ull * 256 * 2);
    _Float16* A1  = (_Float16*)alloc(2560ull * 512 * 2);
    _Float16* h2  = (_Float16*)alloc(2560ull * 256 * 2);
    _Float16* A2  = (_Float16*)alloc(512ull * 512 * 2);

    // Weight / bias prep (tiny)
    pack_wt_kernel<<<(256 * 2048 + 255) / 256, 256, 0, stream>>>(Wself0, Wneigh0, W0t, 1024, 256, 256);
    pack_wt_kernel<<<(256 * 512  + 255) / 256, 256, 0, stream>>>(Wself1, Wneigh1, W1t, 256, 256, 256);
    pack_wt_kernel<<<(64  * 512  + 255) / 256, 256, 0, stream>>>(Wself2, Wneigh2, W2t, 256, 19, 64);
    pad_bias_kernel<<<1, 64, 0, stream>>>(b2, b2p, 19, 64);

    // Layer 0: gather+mean (HBM-bound) then WMMA GEMM [25600,2048]x[2048,256]
    build_a0_kernel<<<25600, 256, 0, stream>>>(x, nbr0, A0);
    {
        dim3 g(25600 / 128, 256 / 64);
        gemm_wmma_kernel<true, true><<<g, 256, 0, stream>>>(A0, W0t, b0, h1, 2048, 256, 256);
    }

    // Layer 1: [2560,512]x[512,256]
    build_a_f16_kernel<<<2560, 64, 0, stream>>>(h1, nbr1, A1, 256, 10, 0.1f);
    {
        dim3 g(2560 / 128, 256 / 64);
        gemm_wmma_kernel<true, true><<<g, 256, 0, stream>>>(A1, W1t, b1, h2, 512, 256, 256);
    }

    // Layer 2: [512,512]x[512,64(pad)] -> write f32 cols < 19
    build_a_f16_kernel<<<512, 64, 0, stream>>>(h2, nbr2, A2, 256, 5, 0.2f);
    {
        dim3 g(512 / 128, 1);
        gemm_wmma_kernel<false, false><<<g, 256, 0, stream>>>(A2, W2t, b2p, (float*)d_out, 512, 19, 19);
    }
}
